// NMSLatentDisentangledStyle_34222299415058
// MI455X (gfx1250) — matrix operand order repaired and checked by reading, not compiled, not run on hardware
//
#include <hip/hip_runtime.h>
#include <math.h>

#define DEV __device__ __forceinline__

typedef __attribute__((ext_vector_type(16))) __bf16 v16bf;
typedef __attribute__((ext_vector_type(8)))  float  v8f;
typedef __attribute__((ext_vector_type(8)))  unsigned short us8;

union BFrag { v16bf v; us8 h[2]; unsigned short u[16]; };

// native f32->bf16 convert (v_cvt_*bf16_f32) instead of bit-twiddled RNE
DEV unsigned short f2bf(float f){
  union { __bf16 b; unsigned short u; } c;
  c.b = (__bf16)f;
  return c.u;
}
DEV float bf2f(unsigned short u){ return __uint_as_float(((unsigned int)u) << 16); }
DEV float sigmf(float x){ return 1.f / (1.f + expf(-x)); }

DEV v8f wmma_bf16(v16bf a, v16bf b, v8f c){
  // D = A(16x32 bf16) x B(32x16 bf16) + C(f32)
  return __builtin_amdgcn_wmma_f32_16x16x32_bf16(false, a, false, b, (short)0, c, false, false);
}

// scalar (SGPR) wave index so tile guards are uniform -> no EXEC masking around WMMA
DEV int wave_id(){ return __builtin_amdgcn_readfirstlane((int)(threadIdx.x >> 5)); }

// ---- fragment loaders (layouts per CDNA5 ISA 7.12.2) ----

// B frag: column n = W row; lanes 0-15 hold K={k0..k0+7, k0+16..k0+23}, lanes 16-31 +8.
// wrow = W + n*Kp (per-lane row pointer), Kp multiple of 32 -> 16B aligned loads.
DEV v16bf load_b_frag(const unsigned short* wrow, int k0){
  int lhi = (threadIdx.x >> 4) & 1;
  const unsigned short* p = wrow + k0 + (lhi << 3);
  BFrag r; r.h[0] = *(const us8*)p; r.h[1] = *(const us8*)(p + 16);
  return r.v;
}

// A frag from LDS bf16 [16 x stride]: lane-group0 K={k0..k0+7, k0+16..k0+23}, group1 +8.
DEV v16bf load_a_lds(const unsigned short* base, int stride, int k0){
  int lane = threadIdx.x & 31;
  int kb = k0 + (((lane >> 4) & 1) << 3);
  const unsigned short* p = base + (lane & 15) * stride + kb;
  BFrag r; r.h[0] = *(const us8*)p; r.h[1] = *(const us8*)(p + 16);
  return r.v;
}

// A frag from a global f32 row, converting to bf16, zero-padding K tail.
// Branch is on the SCALAR condition (k0+32<=K) so the hot path is branch-free.
DEV v16bf load_a_f32(const float* rowp, int k0, int K){
  int lane = threadIdx.x & 31;
  int kb = k0 + (((lane >> 4) & 1) << 3);
  BFrag r;
  if (k0 + 32 <= K){                               // uniform fast path
    #pragma unroll
    for (int i = 0; i < 8; i++){ r.u[i] = f2bf(rowp[kb+i]); r.u[8+i] = f2bf(rowp[kb+16+i]); }
  } else {                                         // tail (last k-step only)
    #pragma unroll
    for (int i = 0; i < 8; i++){
      int k1 = kb + i, k2 = kb + 16 + i;
      r.u[i]   = (k1 < K) ? f2bf(rowp[k1]) : (unsigned short)0;
      r.u[8+i] = (k2 < K) ? f2bf(rowp[k2]) : (unsigned short)0;
    }
  }
  return r.v;
}

// ====================== weight f32 -> bf16 (K padded to 32) ======================
__global__ void cvt_w_k(const float* __restrict__ W, unsigned short* __restrict__ out,
                        int K, int Kp, long total){
  long i = (long)blockIdx.x * blockDim.x + threadIdx.x;
  if (i >= total) return;
  long n = i / Kp; int k = (int)(i - n * Kp);
  out[i] = (k < K) ? f2bf(W[n * (long)K + k]) : (unsigned short)0;
}

// ====================== generic GEMM: C = act(A @ W^T + b1 + b2) ======================
// A: [M,K] f32 (row stride lda). W: [N,Kp] bf16 row-major (gate rows, K padded).
// Block tile 64(M) x 128(N), 8 waves: wave = wm(0..3) x wn(0..1), each wave 1 Mtile x 4 Ntiles.
// k-loop is branch-free: edge tiles are column-clamped (recompute a valid dup tile) and
// discarded at store time, so accumulators never cross a branch.
__global__ __launch_bounds__(256) void gemm_bf16w(
    const float* __restrict__ A, long lda,
    const unsigned short* __restrict__ W, int Kp,
    const float* __restrict__ b1, const float* __restrict__ b2,
    float* __restrict__ C, long ldc,
    int M, int N, int K, int act)
{
  int lane = threadIdx.x & 31;
  int l15 = lane & 15;
  int wu = wave_id();
  int wm = wu & 3, wn = wu >> 2;
  long mbase = (long)blockIdx.y * 64 + wm * 16;
  int  nbase = blockIdx.x * 128 + wn * 64;
  if (mbase >= M || nbase >= N) return;                 // scalar condition
  int jn = (N - nbase + 15) >> 4; if (jn > 4) jn = 4;   // valid tiles (store-time only)
  int nmax = N - 16;
  int c0 = nbase;          if (c0 > nmax) c0 = nmax;    // clamped compute columns
  int c1 = nbase + 16;     if (c1 > nmax) c1 = nmax;
  int c2 = nbase + 32;     if (c2 > nmax) c2 = nmax;
  int c3 = nbase + 48;     if (c3 > nmax) c3 = nmax;
  const unsigned short* w0 = W + (size_t)(c0 + l15) * Kp;
  const unsigned short* w1 = W + (size_t)(c1 + l15) * Kp;
  const unsigned short* w2 = W + (size_t)(c2 + l15) * Kp;
  const unsigned short* w3 = W + (size_t)(c3 + l15) * Kp;

  v8f acc0, acc1, acc2, acc3;
  #pragma unroll
  for (int r = 0; r < 8; r++){ acc0[r] = 0.f; acc1[r] = 0.f; acc2[r] = 0.f; acc3[r] = 0.f; }

  const float* arow = A + (mbase + l15) * lda;
  for (int k0 = 0; k0 < K; k0 += 32){
    v16bf a = load_a_f32(arow, k0, K);
    acc0 = wmma_bf16(a, load_b_frag(w0, k0), acc0);
    acc1 = wmma_bf16(a, load_b_frag(w1, k0), acc1);
    acc2 = wmma_bf16(a, load_b_frag(w2, k0), acc2);
    acc3 = wmma_bf16(a, load_b_frag(w3, k0), acc3);
  }

  int lhi = (lane >> 4) & 1;
  long mrow = mbase + (lhi << 3);
  #pragma unroll
  for (int j = 0; j < 4; j++){
    if (j < jn){                                        // scalar store guard
      v8f acc = (j == 0) ? acc0 : (j == 1) ? acc1 : (j == 2) ? acc2 : acc3;
      int col = nbase + j * 16 + l15;
      float bias = (b1 ? b1[col] : 0.f) + (b2 ? b2[col] : 0.f);
      #pragma unroll
      for (int r = 0; r < 8; r++){
        float v = acc[r] + bias;
        if (act == 1) v = fmaxf(v, 0.f);
        else if (act == 2) v = expf(v);
        else if (act == 3) v = sigmf(v);
        C[(mrow + r) * ldc + col] = v;
      }
    }
  }
}

// ====================== persistent LSTM scan ======================
// grid=(B/16, 2): blockIdx.x = 16-row batch chunk, blockIdx.y = direction (0 fwd, 1 bwd).
// xg: [B*T, 4*Hh] precomputed x@Wih^T + bih + bhh.  Whh: [4*Hh, Hh] bf16.
// Gate order i,f,g,o. Writes h at original time index -> backward output is already re-reversed.
__global__ __launch_bounds__(256) void lstm_scan(
    const float* __restrict__ xgF, const float* __restrict__ xgB,
    const unsigned short* __restrict__ WhhF, const unsigned short* __restrict__ WhhB,
    float* __restrict__ outH, int ldo, int ocolF, int ocolB,
    int Hh, int T)
{
  const int dir = blockIdx.y;
  const float* xg = dir ? xgB : xgF;
  const unsigned short* Whh = dir ? WhhB : WhhF;
  const int ocol = dir ? ocolB : ocolF;

  __shared__ __align__(16) unsigned short h_bf[16 * 128];
  __shared__ float cst[16 * 128];
  __shared__ float gbuf[16 * 512];

  int tid = threadIdx.x, lane = tid & 31;
  int wu = wave_id();
  int G = 4 * Hh, ntiles = G >> 4, ksteps = Hh >> 5;
  long bbase = (long)blockIdx.x * 16;

  for (int i = tid; i < 16 * Hh; i += 256){ h_bf[i] = 0; cst[i] = 0.f; }
  __syncthreads();

  for (int s = 0; s < T; s++){
    int t = dir ? (T - 1 - s) : s;
    // gates += h @ Whh^T  (M=16, N=4Hh, K=Hh); nt loop is scalar-uniform
    for (int nt = wu; nt < ntiles; nt += 8){
      v8f acc;
      #pragma unroll
      for (int r = 0; r < 8; r++) acc[r] = 0.f;
      const unsigned short* wr = Whh + (size_t)(nt * 16 + (lane & 15)) * Hh;
      for (int kk = 0; kk < ksteps; kk++)
        acc = wmma_bf16(load_a_lds(h_bf, Hh, kk * 32), load_b_frag(wr, kk * 32), acc);
      int col = nt * 16 + (lane & 15);
      int mrow = ((lane >> 4) & 1) * 8;
      #pragma unroll
      for (int r = 0; r < 8; r++) gbuf[(mrow + r) * G + col] = acc[r];
    }
    __syncthreads();
    // fused gate nonlinearity + state update + output store
    for (int e = tid; e < 16 * Hh; e += 256){
      int m = e / Hh, j = e - m * Hh;
      long b = bbase + m;
      const float* xr = xg + ((size_t)b * T + t) * G;
      const float* gr = gbuf + m * G;
      float ii = sigmf(gr[j]          + xr[j]);
      float ff = sigmf(gr[Hh + j]     + xr[Hh + j]);
      float gg = tanhf(gr[2 * Hh + j] + xr[2 * Hh + j]);
      float oo = sigmf(gr[3 * Hh + j] + xr[3 * Hh + j]);
      float c = ff * cst[e] + ii * gg;
      cst[e] = c;
      float h = oo * tanhf(c);
      outH[((size_t)b * T + t) * ldo + ocol + j] = h;
      h_bf[e] = f2bf(h);
    }
    __syncthreads();
  }
}

// ====================== persistent dual GRU decode loop ======================
// grid=(2): block 0 = art GRU, block 1 = dyn GRU. B=64, H=64, gates=192, gate order r,z,n.
__global__ __launch_bounds__(256) void gru_decode(
    const float* WihA, const unsigned short* WhhA, const float* bihA, const float* bhhA,
    const float* pWA, const float* pbA, const float* h0A, float* outA,
    const float* WihD, const unsigned short* WhhD, const float* bihD, const float* bhhD,
    const float* pWD, const float* pbD, const float* h0D, float* outD,
    int T)
{
  const bool art = (blockIdx.x == 0);
  const float* Wih = art ? WihA : WihD;
  const unsigned short* Whh = art ? WhhA : WhhD;
  const float* bih = art ? bihA : bihD;
  const float* bhh = art ? bhhA : bhhD;
  const float* pW  = art ? pWA  : pWD;
  const float* pb  = art ? pbA  : pbD;
  const float* h0  = art ? h0A  : h0D;
  float* out       = art ? outA : outD;

  __shared__ __align__(16) unsigned short h_bf[64 * 64];
  __shared__ float gbuf[64 * 192];
  __shared__ float oas[64 * 2];
  __shared__ int   idx[64];

  int tid = threadIdx.x, lane = tid & 31;
  int wu = wave_id();
  int hb = tid >> 2, jb = (tid & 3) << 4;       // each thread owns h[hb][jb..jb+15]
  float hreg[16];
  #pragma unroll
  for (int k = 0; k < 16; k++){
    float v = h0[hb * 64 + jb + k];
    hreg[k] = v;
    h_bf[hb * 64 + jb + k] = f2bf(v);
  }
  __syncthreads();

  for (int s = 0; s < T; s++){
    // gh = h @ Whh^T : M=64, N=192, K=64 -> 48 tiles over 8 waves, fixed 6 tiles/wave
    #pragma unroll
    for (int q = 0; q < 6; q++){
      int tt = q * 8 + wu;                       // scalar
      int mt = tt & 3, nt = tt >> 2;
      v8f acc;
      #pragma unroll
      for (int r = 0; r < 8; r++) acc[r] = 0.f;
      const unsigned short* wr = Whh + (size_t)(nt * 16 + (lane & 15)) * 64;
      #pragma unroll
      for (int kk = 0; kk < 2; kk++)
        acc = wmma_bf16(load_a_lds(h_bf + mt * 16 * 64, 64, kk * 32), load_b_frag(wr, kk * 32), acc);
      int col = nt * 16 + (lane & 15);
      int mrow = mt * 16 + ((lane >> 4) & 1) * 8;
      #pragma unroll
      for (int r = 0; r < 8; r++) gbuf[(mrow + r) * 192 + col] = acc[r];
    }
    __syncthreads();
    // GRU cell; one-hot xin -> gi = bih + Wih[:, idx] (zero xin on first step)
    #pragma unroll
    for (int k = 0; k < 16; k++){
      int j = jb + k, b = hb;
      float xi = (s > 0) ? 1.f : 0.f;
      int ci = (s > 0) ? idx[b] : 0;
      float gir = bih[j]        + xi * Wih[(j)       * 2 + ci];
      float giz = bih[64 + j]   + xi * Wih[(64 + j)  * 2 + ci];
      float gin = bih[128 + j]  + xi * Wih[(128 + j) * 2 + ci];
      float ghr = gbuf[b * 192 + j]        + bhh[j];
      float ghz = gbuf[b * 192 + 64 + j]   + bhh[64 + j];
      float ghn = gbuf[b * 192 + 128 + j]  + bhh[128 + j];
      float r = sigmf(gir + ghr);
      float z = sigmf(giz + ghz);
      float n = tanhf(gin + r * ghn);
      float h = (1.f - z) * n + z * hreg[k];
      hreg[k] = h;
      h_bf[b * 64 + j] = f2bf(h);
    }
    __syncthreads();
    // output head o = h @ pW^T + pb (N=2)
    if (tid < 128){
      int b = tid >> 1, j = tid & 1;
      float sum = pb[j];
      for (int d = 0; d < 64; d++) sum += bf2f(h_bf[b * 64 + d]) * pW[j * 64 + d];
      oas[b * 2 + j] = sum;
      out[((size_t)b * T + s) * 2 + j] = sum;
    }
    __syncthreads();
    if (tid < 64) idx[tid] = (oas[tid * 2 + 1] > oas[tid * 2 + 0]) ? 1 : 0;
    __syncthreads();
  }
}

// ====================== small elementwise kernels ======================
__global__ void sample_z_k(const float* __restrict__ mu, const float* __restrict__ var,
                           const float* __restrict__ eps, float* __restrict__ zout,
                           float* __restrict__ zlst, int ldl, int col, int D, long total){
  long i = (long)blockIdx.x * blockDim.x + threadIdx.x;
  if (i >= total) return;
  long r = i / D; int d = (int)(i - r * D);
  float v = mu[i] + var[i] * eps[i];
  zout[i] = v;
  if (zlst) zlst[r * (long)ldl + col + d] = v;
}

__global__ void qyx_kernel(const float* __restrict__ z, const float* __restrict__ mu,
                           const float* __restrict__ lv, float* __restrict__ probs,
                           float* __restrict__ logits, int M, int Kc, int D){
  int r = blockIdx.x * blockDim.x + threadIdx.x;
  if (r >= M) return;
  const float LOG2PI = 1.8378770664093453f;
  const float* zr = z + (size_t)r * D;
  float lg[4];
  float lpk = logf(1.f / (float)Kc);
  for (int k = 0; k < Kc; k++){
    float s = 0.f;
    const float* muk = mu + k * D;
    const float* lvk = lv + k * D;
    for (int d = 0; d < D; d++){
      float df = zr[d] - muk[d];
      s += df * df * expf(-lvk[d]) + lvk[d] + LOG2PI;
    }
    lg[k] = -0.5f * s + lpk;
  }
  float mx = lg[0];
  for (int k = 1; k < Kc; k++) mx = fmaxf(mx, lg[k]);
  float den = 0.f;
  for (int k = 0; k < Kc; k++) den += expf(lg[k] - mx);
  for (int k = 0; k < Kc; k++){
    probs[(size_t)r * Kc + k] = expf(lg[k] - mx) / den;
    if (logits) logits[(size_t)r * Kc + k] = lg[k];
  }
}

__global__ void concat_feat(const float* __restrict__ pr, const float* __restrict__ zlst,
                            float* __restrict__ feat, long total){
  long i = (long)blockIdx.x * blockDim.x + threadIdx.x;
  if (i >= total) return;
  long r = i / 216; int c = (int)(i - r * 216);
  feat[i] = (c < 88) ? pr[r * 88 + c] : zlst[r * 128 + (c - 88)];
}

// ====================== host orchestration ======================
extern "C" void kernel_launch(void* const* d_in, const int* in_sizes, int n_in,
                              void* d_out, int out_size, void* d_ws, size_t ws_size,
                              hipStream_t stream) {
  (void)in_sizes; (void)n_in; (void)out_size; (void)ws_size;
  const int B = 64, T = 625, BT = B * T, Z = 64, H = 256, Hh = 128;

  auto IN = [&](int i)->const float* { return (const float*)d_in[i]; };

  // deterministic bump allocator over d_ws
  char* wsb = (char*)d_ws;
  size_t off = 0;
  auto alloc = [&](size_t bytes)->void* {
    void* p = wsb + off;
    off = (off + bytes + 255) & ~(size_t)255;
    return p;
  };

  // ---- convert all matmul weights to bf16 (K padded to mult of 32) ----
  auto cvt = [&](int idx, int N, int K)->unsigned short* {
    int Kp = (K + 31) & ~31;
    long total = (long)N * Kp;
    unsigned short* p = (unsigned short*)alloc((size_t)total * 2);
    cvt_w_k<<<dim3((unsigned)((total + 255) / 256)), dim3(256), 0, stream>>>(IN(idx), p, K, Kp, total);
    return p;
  };
  unsigned short* w_conv   = cvt(5, 256, 80);     // Kp=96
  unsigned short* w_ef_ih  = cvt(7, 512, 256);
  unsigned short* w_ef_hh  = cvt(8, 512, 128);
  unsigned short* w_eb_ih  = cvt(11, 512, 256);
  unsigned short* w_eb_hh  = cvt(12, 512, 128);
  unsigned short* w_mua    = cvt(15, 64, 256);
  unsigned short* w_vara   = cvt(17, 64, 256);
  unsigned short* w_mud    = cvt(19, 64, 256);
  unsigned short* w_vard   = cvt(21, 64, 256);
  unsigned short* w_sf_ih  = cvt(23, 128, 128);
  unsigned short* w_sf_hh  = cvt(24, 128, 32);
  unsigned short* w_sb_ih  = cvt(27, 128, 128);
  unsigned short* w_sb_hh  = cvt(28, 128, 32);
  unsigned short* w_mus    = cvt(31, 64, 64);
  unsigned short* w_vars   = cvt(33, 64, 64);
  unsigned short* w_z2a    = cvt(35, 64, 64);
  unsigned short* w_z2d    = cvt(37, 64, 64);
  unsigned short* w_gA_hh  = cvt(40, 192, 64);
  unsigned short* w_gD_hh  = cvt(44, 192, 64);
  unsigned short* w_d0f_ih = cvt(51, 512, 216);   // Kp=224
  unsigned short* w_d0f_hh = cvt(52, 512, 128);
  unsigned short* w_d0b_ih = cvt(55, 512, 216);
  unsigned short* w_d0b_hh = cvt(56, 512, 128);
  unsigned short* w_d1f_ih = cvt(59, 512, 256);
  unsigned short* w_d1f_hh = cvt(60, 512, 128);
  unsigned short* w_d1b_ih = cvt(63, 512, 256);
  unsigned short* w_d1b_hh = cvt(64, 512, 128);
  unsigned short* w_oc     = cvt(67, 256, 256);
  unsigned short* w_oc2    = cvt(69, 80, 256);

  // ---- float scratch (aggressively reused) ----
  float* F0 = (float*)alloc((size_t)BT * 256 * 4); // conv out | style out | feat | dec1 out
  float* F1 = (float*)alloc((size_t)BT * 512 * 4); // xg forward (all layers)
  float* F2 = (float*)alloc((size_t)BT * 512 * 4); // xg backward
  float* F3 = (float*)alloc((size_t)BT * 256 * 4); // enc q | dec0 out | out_conv out
  float* F4 = (float*)alloc((size_t)BT * 128 * 4); // z_lst
  float* S3 = (float*)alloc((size_t)B * Z * 4);    // GRU h0 art
  float* S4 = (float*)alloc((size_t)B * Z * 4);    // GRU h0 dyn

  // ---- d_out slices (reference return order, all f32) ----
  float* o = (float*)d_out;
  float* o_xhat = o;                    // [64,625,80]
  float* o_za   = o_xhat + 3200000;     // [64,625,64]
  float* o_artc = o_za   + 2560000;     // [64,625,2]
  float* o_mua  = o_artc + 80000;
  float* o_vara = o_mua  + 2560000;
  float* o_zd   = o_vara + 2560000;
  float* o_dync = o_zd   + 2560000;
  float* o_mud  = o_dync + 80000;
  float* o_vard = o_mud  + 2560000;
  float* o_zs   = o_vard + 2560000;     // [64,64]
  float* o_mus  = o_zs   + 4096;
  float* o_vars = o_mus  + 4096;
  float* o_logs = o_vars + 4096;        // [64,4]
  float* o_probs= o_logs + 256;
  float* o_arth = o_probs+ 256;         // [64,625,2]
  float* o_dynh = o_arth + 80000;

  auto gemm = [&](const float* A, long lda, const unsigned short* W, int Kp,
                  const float* b1, const float* b2, float* C, long ldc,
                  int M, int N, int K, int act){
    dim3 g((unsigned)((N + 127) / 128), (unsigned)((M + 63) / 64));
    gemm_bf16w<<<g, dim3(256), 0, stream>>>(A, lda, W, Kp, b1, b2, C, ldc, M, N, K, act);
  };

  // 1) conv_enc: q1 = relu(x @ Wc^T + bc)
  gemm(IN(0), 80, w_conv, 96, IN(6), nullptr, F0, 256, BT, H, 80, 1);
  // 2) encoder LSTM input gates (bih+bhh folded in)
  gemm(F0, 256, w_ef_ih, 256, IN(9),  IN(10), F1, 512, BT, 512, 256, 0);
  gemm(F0, 256, w_eb_ih, 256, IN(13), IN(14), F2, 512, BT, 512, 256, 0);
  // 3) encoder BiLSTM scan -> q in F3 (fwd cols 0..127, bwd 128..255)
  lstm_scan<<<dim3(4, 2), dim3(256), 0, stream>>>(F1, F2, w_ef_hh, w_eb_hh, F3, 256, 0, 128, Hh, T);
  // 4) heads
  gemm(F3, 256, w_mua,  256, IN(16), nullptr, o_mua,  64, BT, 64, 256, 0);
  gemm(F3, 256, w_vara, 256, IN(18), nullptr, o_vara, 64, BT, 64, 256, 2);
  gemm(F3, 256, w_mud,  256, IN(20), nullptr, o_mud,  64, BT, 64, 256, 0);
  gemm(F3, 256, w_vard, 256, IN(22), nullptr, o_vard, 64, BT, 64, 256, 2);
  // 5) sample z_a, z_d (also build z_lst in F4) + per-frame GMM posteriors
  {
    long tot = (long)BT * Z;
    unsigned g = (unsigned)((tot + 255) / 256);
    sample_z_k<<<g, 256, 0, stream>>>(o_mua, o_vara, IN(2), o_za, F4, 128, 0,  Z, tot);
    sample_z_k<<<g, 256, 0, stream>>>(o_mud, o_vard, IN(3), o_zd, F4, 128, 64, Z, tot);
    unsigned gr = (unsigned)((BT + 255) / 256);
    qyx_kernel<<<gr, 256, 0, stream>>>(o_za, IN(71), IN(72), o_artc, nullptr, BT, 2, Z);
    qyx_kernel<<<gr, 256, 0, stream>>>(o_zd, IN(73), IN(74), o_dync, nullptr, BT, 2, Z);
  }
  // 6) style BiLSTM (hidden 32); outputs to F0 as [B*T,64]
  gemm(F4, 128, w_sf_ih, 128, IN(25), IN(26), F1, 128, BT, 128, 128, 0);
  gemm(F4, 128, w_sb_ih, 128, IN(29), IN(30), F2, 128, BT, 128, 128, 0);
  lstm_scan<<<dim3(4, 2), dim3(256), 0, stream>>>(F1, F2, w_sf_hh, w_sb_hh, F0, 64, 0, 32, 32, T);
  // 7) style heads on qs = bilstm[:, -1] (row t=T-1, row stride T*64)
  gemm(F0 + (size_t)(T - 1) * 64, (long)T * 64, w_mus,  64, IN(32), nullptr, o_mus,  64, B, 64, 64, 0);
  gemm(F0 + (size_t)(T - 1) * 64, (long)T * 64, w_vars, 64, IN(34), nullptr, o_vars, 64, B, 64, 64, 0);
  {
    long tot = (long)B * Z;
    sample_z_k<<<dim3((unsigned)((tot + 255) / 256)), 256, 0, stream>>>(
        o_mus, o_vars, IN(4), o_zs, nullptr, 64, 0, Z, tot);
    qyx_kernel<<<1, 256, 0, stream>>>(o_zs, IN(75), IN(76), o_probs, o_logs, B, 4, Z);
  }
  // 8) GRU decode (argmax one-hot feedback, 625 steps, both GRUs concurrently)
  gemm(o_zs, 64, w_z2a, 64, IN(36), nullptr, S3, 64, B, 64, 64, 0);
  gemm(o_zs, 64, w_z2d, 64, IN(38), nullptr, S4, 64, B, 64, 64, 0);
  gru_decode<<<dim3(2), dim3(256), 0, stream>>>(
      IN(39), w_gA_hh, IN(41), IN(42), IN(47), IN(48), S3, o_arth,
      IN(43), w_gD_hh, IN(45), IN(46), IN(49), IN(50), S4, o_dynh, T);
  // 9) decoder: feat = [pr, z_lst] -> 2-layer BiLSTM -> heads
  {
    long ftot = (long)BT * 216;
    concat_feat<<<dim3((unsigned)((ftot + 255) / 256)), 256, 0, stream>>>(IN(1), F4, F0, ftot);
  }
  gemm(F0, 216, w_d0f_ih, 224, IN(53), IN(54), F1, 512, BT, 512, 216, 0);
  gemm(F0, 216, w_d0b_ih, 224, IN(57), IN(58), F2, 512, BT, 512, 216, 0);
  lstm_scan<<<dim3(4, 2), dim3(256), 0, stream>>>(F1, F2, w_d0f_hh, w_d0b_hh, F3, 256, 0, 128, Hh, T);
  gemm(F3, 256, w_d1f_ih, 256, IN(61), IN(62), F1, 512, BT, 512, 256, 0);
  gemm(F3, 256, w_d1b_ih, 256, IN(65), IN(66), F2, 512, BT, 512, 256, 0);
  lstm_scan<<<dim3(4, 2), dim3(256), 0, stream>>>(F1, F2, w_d1f_hh, w_d1b_hh, F0, 256, 0, 128, Hh, T);
  gemm(F0, 256, w_oc,  256, IN(68), nullptr, F3, 256, BT, 256, 256, 1);      // relu
  gemm(F3, 256, w_oc2, 256, IN(70), nullptr, o_xhat, 80, BT, 80, 256, 3);    // sigmoid
}
